// model_ablation_no_cau_24489903522215
// MI455X (gfx1250) — compile-verified, matrix operand order checked by reading, not run on hardware
//
#include <hip/hip_runtime.h>
#include <hip/hip_bf16.h>
#include <math.h>

// ---------------------------------------------------------------------------
// HGT forward for MI455X (gfx1250, wave32, WMMA).
// Dense math: v_wmma_f32_16x16x32_f16, 16x128 output tile per wave,
// B operands pre-packed into per-lane WMMA fragment layout (f16, L2-resident).
// ---------------------------------------------------------------------------

typedef __attribute__((ext_vector_type(16))) _Float16 v16h;
typedef __attribute__((ext_vector_type(8)))  float    v8f;

#define HID    128
#define HEADS  8
#define DK     16
#define NDOC   5000

// order-preserving float<->uint encoding for atomicMax on floats
static __device__ __forceinline__ unsigned fenc(float f) {
  unsigned u = __float_as_uint(f);
  return (u & 0x80000000u) ? ~u : (u | 0x80000000u);
}
static __device__ __forceinline__ float fdec(unsigned u) {
  return (u & 0x80000000u) ? __uint_as_float(u & 0x7FFFFFFFu) : __uint_as_float(~u);
}

// ---------------------------------------------------------------------------
// Pack W[K x 128] (f32, row-major) into WMMA B-fragment layout (f16):
//   Wp[((kt*8 + ct)*32 + lane)*16 + e],  e: k = kt*32 + kb + (e<8 ? e : e+8),
//   kb = (lane<16)?0:8, n = ct*16 + (lane&15).  Zero-padded for k >= K.
// ---------------------------------------------------------------------------
__global__ void hgt_pack_wfrag(const float* __restrict__ W, int K,
                               _Float16* __restrict__ Wp, int total) {
  int i = blockIdx.x * blockDim.x + threadIdx.x;
  if (i >= total) return;
  int e    = i & 15;
  int lane = (i >> 4) & 31;
  int ct   = (i >> 9) & 7;
  int kt   = i >> 12;
  int kb = (lane < 16) ? 0 : 8;
  int k = kt * 32 + kb + (e < 8 ? e : e + 8);
  int n = ct * 16 + (lane & 15);
  Wp[i] = (k < K) ? (_Float16)W[(long)k * HID + n] : (_Float16)0.f;
}

// Pack msg[8][16][16] into fragment layout Mp[(h*32 + lane)*16 + e], zeros k>=16.
__global__ void hgt_pack_msgfrag(const float* __restrict__ msgW,
                                 _Float16* __restrict__ Mp) {
  int i = blockIdx.x * blockDim.x + threadIdx.x;   // 8*32*16 = 4096
  if (i >= HEADS * 32 * 16) return;
  int e    = i & 15;
  int lane = (i >> 4) & 31;
  int h    = i >> 9;
  int kb = (lane < 16) ? 0 : 8;
  int k = kb + (e < 8 ? e : e + 8);
  int n = lane & 15;
  Mp[i] = (k < DK) ? (_Float16)msgW[h * DK * DK + k * DK + n] : (_Float16)0.f;
}

// ---------------------------------------------------------------------------
// C[M x 128] = gather(A)[M x K] @ W[K x 128] + bias.
// One wave computes a full 16x128 row block: A fragment loaded once per K-step
// (4 x float4), 8 WMMAs against packed B fragments.
// ---------------------------------------------------------------------------
__global__ __launch_bounds__(32)
void hgt_gemm_wmma(const float* __restrict__ A, const int* __restrict__ gidx,
                   int M, int K, int lda,
                   const _Float16* __restrict__ Wp, const float* __restrict__ bias,
                   float* __restrict__ C) {
  int lane = threadIdx.x;
  int r0 = blockIdx.x * 16;
  int m = lane & 15;
  int rowc = r0 + m; if (rowc > M - 1) rowc = M - 1;       // clamp: no divergence
  long ar = gidx ? (long)gidx[rowc] : (long)rowc;
  const float* Arow = A + ar * (long)lda;
  int kb = (lane < 16) ? 0 : 8;

  v8f acc[8] = {};
  int Kfull = K & ~31;
  int kt = 0;
  for (; kt * 32 < Kfull; ++kt) {
    const float* p = Arow + kt * 32 + kb;
    __builtin_prefetch(p + 64, 0, 1);
    float4 f0 = *(const float4*)(p);
    float4 f1 = *(const float4*)(p + 4);
    float4 f2 = *(const float4*)(p + 16);
    float4 f3 = *(const float4*)(p + 20);
    v16h a;
    a[0]=(_Float16)f0.x;  a[1]=(_Float16)f0.y;  a[2]=(_Float16)f0.z;  a[3]=(_Float16)f0.w;
    a[4]=(_Float16)f1.x;  a[5]=(_Float16)f1.y;  a[6]=(_Float16)f1.z;  a[7]=(_Float16)f1.w;
    a[8]=(_Float16)f2.x;  a[9]=(_Float16)f2.y;  a[10]=(_Float16)f2.z; a[11]=(_Float16)f2.w;
    a[12]=(_Float16)f3.x; a[13]=(_Float16)f3.y; a[14]=(_Float16)f3.z; a[15]=(_Float16)f3.w;
    const _Float16* bp = Wp + (long)kt * 4096 + lane * 16;
#pragma unroll
    for (int ct = 0; ct < 8; ++ct) {
      v16h b = *(const v16h*)(bp + ct * 512);
      acc[ct] = __builtin_amdgcn_wmma_f32_16x16x32_f16(false, a, false, b, (short)0,
                                                       acc[ct], false, false);
    }
  }
  if (Kfull < K) {                                         // single branchless tail
    v16h a;
#pragma unroll
    for (int e = 0; e < 16; ++e) {
      int k = Kfull + kb + (e < 8 ? e : e + 8);
      float va = Arow[k < K ? k : 0];                      // clamped addr, no branch
      a[e] = (_Float16)(k < K ? va : 0.f);
    }
    const _Float16* bp = Wp + (long)kt * 4096 + lane * 16;
#pragma unroll
    for (int ct = 0; ct < 8; ++ct) {
      v16h b = *(const v16h*)(bp + ct * 512);
      acc[ct] = __builtin_amdgcn_wmma_f32_16x16x32_f16(false, a, false, b, (short)0,
                                                       acc[ct], false, false);
    }
  }

  int rbase = r0 + ((lane < 16) ? 0 : 8);
  if (r0 + 16 <= M) {                                      // wave-uniform fast path
#pragma unroll
    for (int ct = 0; ct < 8; ++ct) {
      int n = ct * 16 + m;
      float bn = bias ? bias[n] : 0.f;
#pragma unroll
      for (int r = 0; r < 8; ++r)
        C[(long)(rbase + r) * HID + n] = acc[ct][r] + bn;
    }
  } else {
#pragma unroll
    for (int ct = 0; ct < 8; ++ct) {
      int n = ct * 16 + m;
      float bn = bias ? bias[n] : 0.f;
#pragma unroll
      for (int r = 0; r < 8; ++r)
        if (rbase + r < M) C[(long)(rbase + r) * HID + n] = acc[ct][r] + bn;
    }
  }
}

// ---------------------------------------------------------------------------
// Per-edge message einsum('ehk,hkd') as WMMA (16 edges x head, K=16 pad 32),
// fused with softmax-weighted atomic scatter into acc[dst].
// ---------------------------------------------------------------------------
__global__ __launch_bounds__(32)
void hgt_msg_scatter_wmma(const float* __restrict__ V, const _Float16* __restrict__ Mp,
                          const int* __restrict__ srcI, const int* __restrict__ dstI,
                          const float* __restrict__ att, const unsigned* __restrict__ menc,
                          const float* __restrict__ ssum, int E,
                          float* __restrict__ acc) {
  int lane = threadIdx.x;
  int e0 = blockIdx.x * 16;
  int h = blockIdx.y;
  int m = lane & 15;
  int ec = e0 + m; if (ec > E - 1) ec = E - 1;             // clamp: no divergence
  const float* Vrow = V + (long)srcI[ec] * HID + h * DK;
  int kb = (lane < 16) ? 0 : 8;
  float4 f0 = *(const float4*)(Vrow + kb);
  float4 f1 = *(const float4*)(Vrow + kb + 4);
  v16h a;
  a[0]=(_Float16)f0.x; a[1]=(_Float16)f0.y; a[2]=(_Float16)f0.z; a[3]=(_Float16)f0.w;
  a[4]=(_Float16)f1.x; a[5]=(_Float16)f1.y; a[6]=(_Float16)f1.z; a[7]=(_Float16)f1.w;
#pragma unroll
  for (int e = 8; e < 16; ++e) a[e] = (_Float16)0.f;       // k>=16 pad
  v16h b = *(const v16h*)(Mp + (h * 32 + lane) * 16);
  v8f c = {};
  c = __builtin_amdgcn_wmma_f32_16x16x32_f16(false, a, false, b, (short)0, c,
                                             false, false);
  int rbase = e0 + ((lane < 16) ? 0 : 8);
#pragma unroll
  for (int r = 0; r < 8; ++r) {
    int ee = rbase + r;
    if (ee < E) {
      int d = dstI[ee];
      float av = att[(long)ee * HEADS + h];
      float mx = fdec(menc[(long)d * HEADS + h]);
      float sv = ssum[(long)d * HEADS + h];
      float coef = __expf(av - mx) / sv;
      atomicAdd(&acc[(long)d * HID + h * DK + m], coef * c[r]);
    }
  }
}

// att[e,h] = sum_d (q[dst,h,d] + w*attW[h,d]) * k[src,h,d] / sqrt(dk); seg-max; deg.
__global__ void hgt_att(const float* __restrict__ q, const float* __restrict__ k,
                        const int* __restrict__ srcI, const int* __restrict__ dstI,
                        const float* __restrict__ w, const float* __restrict__ attW,
                        long EH, float* __restrict__ att,
                        unsigned* __restrict__ menc, unsigned* __restrict__ deg) {
  long i = (long)blockIdx.x * blockDim.x + threadIdx.x;
  if (i >= EH) return;
  int e = (int)(i >> 3), h = (int)(i & 7);
  int s = srcI[e], d = dstI[e];
  float we = w[e];
  const float4* qr = (const float4*)(q + (long)d * HID + h * DK);
  const float4* kr = (const float4*)(k + (long)s * HID + h * DK);
  const float4* aw = (const float4*)(attW + h * DK);
  float sc = 0.f;
#pragma unroll
  for (int t = 0; t < 4; ++t) {
    float4 qq = qr[t], kk = kr[t], aa = aw[t];
    sc += (qq.x + we * aa.x) * kk.x + (qq.y + we * aa.y) * kk.y +
          (qq.z + we * aa.z) * kk.z + (qq.w + we * aa.w) * kk.w;
  }
  sc *= 0.25f;                                             // 1/sqrt(16)
  att[i] = sc;
  atomicMax(&menc[(long)d * HEADS + h], fenc(sc));
  if (h == 0) atomicAdd(&deg[d], 1u);
}

__global__ void hgt_expsum(const int* __restrict__ dstI, const float* __restrict__ att,
                           const unsigned* __restrict__ menc, float* __restrict__ s, long EH) {
  long i = (long)blockIdx.x * blockDim.x + threadIdx.x;
  if (i >= EH) return;
  int e = (int)(i >> 3), h = (int)(i & 7);
  int d = dstI[e];
  atomicAdd(&s[(long)d * HEADS + h], __expf(att[i] - fdec(menc[(long)d * HEADS + h])));
}

__global__ void hgt_cnt(const unsigned* __restrict__ deg, float* __restrict__ cnt, int M) {
  int i = blockIdx.x * blockDim.x + threadIdx.x;
  if (i < M) cnt[i] += deg[i] ? 1.f : 0.f;
}

// x = relu(acc / max(cnt,1)); LayerNorm(128) -> h. One block (128 thr) per node.
__global__ __launch_bounds__(HID)
void hgt_finalize_ln(const float* __restrict__ acc, const float* __restrict__ cnt,
                     const float* __restrict__ g, const float* __restrict__ b,
                     float* __restrict__ h) {
  __shared__ float red[HID];
  int n = blockIdx.x, t = threadIdx.x;
  float c = fmaxf(cnt[n], 1.f);
  float x = fmaxf(acc[(long)n * HID + t] / c, 0.f);
  red[t] = x; __syncthreads();
  for (int o = HID / 2; o > 0; o >>= 1) { if (t < o) red[t] += red[t + o]; __syncthreads(); }
  float mu = red[0] * (1.f / HID); __syncthreads();
  float xc = x - mu;
  red[t] = xc * xc; __syncthreads();
  for (int o = HID / 2; o > 0; o >>= 1) { if (t < o) red[t] += red[t + o]; __syncthreads(); }
  float var = red[0] * (1.f / HID);
  h[(long)n * HID + t] = xc * rsqrtf(var + 1e-5f) * g[t] + b[t];
}

__global__ void hgt_tile_doc(const float* __restrict__ dg, float* __restrict__ hd, long total) {
  long i = (long)blockIdx.x * blockDim.x + threadIdx.x;
  if (i < total) hd[i] = dg[i & (HID - 1)];
}
__global__ void hgt_gather_rows(const float* __restrict__ emb, const int* __restrict__ idx,
                                float* __restrict__ out, long total) {
  long i = (long)blockIdx.x * blockDim.x + threadIdx.x;
  if (i < total) out[i] = emb[(long)idx[i >> 7] * HID + (i & (HID - 1))];
}

#define POOL_ROWS 128
__global__ __launch_bounds__(HID)
void hgt_pool_max(const float* __restrict__ h, int M, const float* __restrict__ skip,
                  unsigned* __restrict__ pool) {
  int t = threadIdx.x;
  float scale = skip ? (1.f / (1.f + __expf(-skip[0]))) : 1.f;
  long r0 = (long)blockIdx.x * POOL_ROWS;
  long r1 = r0 + POOL_ROWS; if (r1 > M) r1 = M;
  float mx = -3.4e38f;
  for (long r = r0; r < r1; ++r) mx = fmaxf(mx, h[r * HID + t] * scale);
  atomicMax(&pool[t], fenc(mx));
}

__global__ void hgt_final_out(const unsigned* __restrict__ pool, const float* __restrict__ outW,
                              const float* __restrict__ outB, const float* __restrict__ y,
                              float* __restrict__ out) {
  if (threadIdx.x != 0) return;
  float logit = outB[0];
  for (int i = 0; i < 3 * HID; ++i) logit += fdec(pool[i]) * outW[i];
  float yy = y[0];
  float loss = fmaxf(logit, 0.f) - logit * yy + log1pf(__expf(-fabsf(logit)));
  out[0] = loss;
  out[1] = 1.f / (1.f + __expf(-logit));
}

// ---------------------------------------------------------------------------
// Host orchestration (graph-capture safe: kernels + hipMemsetAsync only).
// ---------------------------------------------------------------------------
extern "C" void kernel_launch(void* const* d_in, const int* in_sizes, int n_in,
                              void* d_out, int out_size, void* d_ws, size_t ws_size,
                              hipStream_t stream) {
  (void)n_in; (void)out_size; (void)ws_size;
  const int*   word_id      = (const int*)d_in[0];
  const int*   topic_id     = (const int*)d_in[1];
  const float* y_data       = (const float*)d_in[20];
  const float* adapt_W      = (const float*)d_in[21];
  const float* adapt_b      = (const float*)d_in[22];
  const float* doc_gen      = (const float*)d_in[23];
  const float* out_W        = (const float*)d_in[96];
  const float* out_b        = (const float*)d_in[97];
  const float* ts_topic     = (const float*)d_in[98];
  const float* ts_word      = (const float*)d_in[99];
  const float* topic_embeds = (const float*)d_in[100];
  const float* word_embeds  = (const float*)d_in[101];

  int Mw = in_sizes[0], Mt = in_sizes[1], Md = NDOC;
  int Kin = in_sizes[21] / HID;                 // 300
  int Mn[3] = { Mw, Mt, Md };                   // 0=word 1=topic 2=doc

  struct ETH { int si, st, dt, es; };
  const ETH ets[6] = { {2,0,2,4}, {5,1,2,1}, {8,1,1,2}, {11,0,0,5}, {14,1,0,3}, {17,2,0,0} };
  const int sArr[3] = { 2, 1, 0 };              // sorted pos: doc<topic<word

  char* wsp = (char*)d_ws; size_t off = 0;
  auto carve = [&](size_t bytes) -> void* {
    void* p = wsp + off; off = (off + bytes + 255) & ~(size_t)255; return p;
  };
  float *h[3], *kb[3], *qb[3], *vb[3], *ab[3], *cb[3];
  for (int t = 0; t < 3; ++t) {
    size_t hb = (size_t)Mn[t] * HID * 4;
    h[t] = (float*)carve(hb); kb[t] = (float*)carve(hb); qb[t] = (float*)carve(hb);
    vb[t] = (float*)carve(hb); ab[t] = (float*)carve(hb);
    cb[t] = (float*)carve((size_t)Mn[t] * 4);
  }
  int Emax = 0;
  for (int i = 0; i < 6; ++i) if (in_sizes[ets[i].si] > Emax) Emax = in_sizes[ets[i].si];
  float*    attb = (float*)carve((size_t)Emax * HEADS * 4);
  unsigned* menc = (unsigned*)carve((size_t)Mw * HEADS * 4);
  float*    sb   = (float*)carve((size_t)Mw * HEADS * 4);
  unsigned* degb = (unsigned*)carve((size_t)Mw * 4);
  unsigned* pool = (unsigned*)carve(3 * HID * 4);
  int KTin = (Kin + 31) / 32;                   // 10 tiles for K=300
  _Float16* Wp = (_Float16*)carve((size_t)KTin * 4096 * 2);   // fits K<=320
  _Float16* Mp = (_Float16*)carve((size_t)HEADS * 32 * 16 * 2);

  auto packW = [&](const float* W, int K) {
    int total = ((K + 31) / 32) * 4096;
    hgt_pack_wfrag<<<(total + 255) / 256, 256, 0, stream>>>(W, K, Wp, total);
  };

  // ---- initial features ----
  {
    long td = (long)Md * HID;
    hgt_tile_doc<<<(unsigned)((td + 255) / 256), 256, 0, stream>>>(doc_gen, h[2], td);
    long tt = (long)Mt * HID;
    hgt_gather_rows<<<(unsigned)((tt + 255) / 256), 256, 0, stream>>>(topic_embeds, topic_id, h[1], tt);
    packW(adapt_W, Kin);
    hgt_gemm_wmma<<<(Mw + 15) / 16, 32, 0, stream>>>(word_embeds, word_id, Mw, Kin, Kin,
                                                     Wp, adapt_b, h[0]);
  }

  // ---- 2 HGT layers ----
  for (int L = 0; L < 2; ++L) {
    int base = 24 + L * 36;
    for (int t = 0; t < 3; ++t) {
      int s = sArr[t];
      unsigned g = (Mn[t] + 15) / 16;
      packW((const float*)d_in[base + s], HID);
      hgt_gemm_wmma<<<g, 32, 0, stream>>>(h[t], nullptr, Mn[t], HID, HID,
          Wp, (const float*)d_in[base + 15 + s], kb[t]);
      packW((const float*)d_in[base + 3 + s], HID);
      hgt_gemm_wmma<<<g, 32, 0, stream>>>(h[t], nullptr, Mn[t], HID, HID,
          Wp, (const float*)d_in[base + 18 + s], qb[t]);
      packW((const float*)d_in[base + 6 + s], HID);
      hgt_gemm_wmma<<<g, 32, 0, stream>>>(h[t], nullptr, Mn[t], HID, HID,
          Wp, (const float*)d_in[base + 21 + s], vb[t]);
      hipMemsetAsync(ab[t], 0, (size_t)Mn[t] * HID * 4, stream);
      hipMemsetAsync(cb[t], 0, (size_t)Mn[t] * 4, stream);
    }
    for (int ei = 0; ei < 6; ++ei) {
      const ETH& et = ets[ei];
      int E = in_sizes[et.si];
      const int*   srcI = (const int*)d_in[et.si];
      const int*   dstI = (const int*)d_in[et.si + 1];
      const float* we   = (const float*)d_in[et.si + 2];
      const float* attW = (const float*)d_in[base + 9 + et.es];
      const float* msgW = (const float*)d_in[base + 30 + et.es];
      int nd = Mn[et.dt];
      hipMemsetAsync(menc, 0, (size_t)nd * HEADS * 4, stream);
      hipMemsetAsync(sb,   0, (size_t)nd * HEADS * 4, stream);
      hipMemsetAsync(degb, 0, (size_t)nd * 4, stream);
      long EH = (long)E * HEADS;
      unsigned bl = (unsigned)((EH + 255) / 256);
      hgt_att<<<bl, 256, 0, stream>>>(qb[et.dt], kb[et.st], srcI, dstI, we, attW, EH,
                                      attb, menc, degb);
      hgt_expsum<<<bl, 256, 0, stream>>>(dstI, attb, menc, sb, EH);
      hgt_pack_msgfrag<<<16, 256, 0, stream>>>(msgW, Mp);
      hgt_msg_scatter_wmma<<<dim3((E + 15) / 16, HEADS), 32, 0, stream>>>(
          vb[et.st], Mp, srcI, dstI, attb, menc, sb, E, ab[et.dt]);
      hgt_cnt<<<(nd + 255) / 256, 256, 0, stream>>>(degb, cb[et.dt], nd);
    }
    for (int t = 0; t < 3; ++t) {
      int s = sArr[t];
      hgt_finalize_ln<<<Mn[t], HID, 0, stream>>>(ab[t], cb[t],
          (const float*)d_in[base + 27 + s],   // ln_g
          (const float*)d_in[base + 24 + s],   // ln_b
          h[t]);
    }
  }

  // ---- pooled max -> logit -> [loss, prob] ----
  hipMemsetAsync(pool, 0, 3 * HID * 4, stream);
  hgt_pool_max<<<(Md + POOL_ROWS - 1) / POOL_ROWS, HID, 0, stream>>>(h[2], Md, nullptr,  pool);
  hgt_pool_max<<<(Mt + POOL_ROWS - 1) / POOL_ROWS, HID, 0, stream>>>(h[1], Mt, ts_topic, pool + HID);
  hgt_pool_max<<<(Mw + POOL_ROWS - 1) / POOL_ROWS, HID, 0, stream>>>(h[0], Mw, ts_word,  pool + 2 * HID);
  hgt_final_out<<<1, 32, 0, stream>>>(pool, out_W, out_b, y_data, (float*)d_out);
}
